// MiniBAKA_25477746000467
// MI455X (gfx1250) — matrix operand order repaired and checked by reading, not compiled, not run on hardware
//
#include <hip/hip_runtime.h>
#include <math.h>

// ---------------- Problem constants ----------------
#define Bz   8
#define Tz   1024
#define Fz   64
#define Dz   512
#define Hz   8
#define DHz  64
#define DFFz 2048
#define Lz   3
#define BT   (Bz*Tz)          // 8192 rows

// ---------------- WMMA vector types ----------------
typedef __attribute__((ext_vector_type(16))) __bf16 v16bf;
typedef __attribute__((ext_vector_type(8)))  __bf16 v8bf;
typedef __attribute__((ext_vector_type(8)))  float  v8f;
typedef __attribute__((ext_vector_type(4)))  int    v4i;

static __device__ __forceinline__ v16bf mk16(v8bf lo, v8bf hi) {
  return __builtin_shufflevector(lo, hi, 0,1,2,3,4,5,6,7,8,9,10,11,12,13,14,15);
}

// ---------------- Async global->LDS (CDNA5) ----------------
typedef __attribute__((address_space(1))) v4i* gp_v4i;   // global v4i*
typedef __attribute__((address_space(3))) v4i* lp_v4i;   // LDS v4i*

#if defined(__has_builtin)
#  if __has_builtin(__builtin_amdgcn_global_load_async_to_lds_b128)
#    define ASYNC_LD_BUILTIN 1
#  endif
#  if __has_builtin(__builtin_amdgcn_s_wait_asynccnt)
#    define ASYNC_WAIT_BUILTIN 1
#  endif
#endif

#ifndef ASYNC_LD_BUILTIN
#warning "gfx1250 async-to-LDS builtin NOT found; using inline-asm global_load_async_to_lds_b128"
#endif
#ifndef ASYNC_WAIT_BUILTIN
#warning "s_wait_asynccnt builtin NOT found; using inline asm"
#endif

static __device__ __forceinline__ void async_ld_b128(const void* g, void* l) {
#if defined(ASYNC_LD_BUILTIN)
  __builtin_amdgcn_global_load_async_to_lds_b128(
      (gp_v4i)(uintptr_t)g, (lp_v4i)(unsigned)(uintptr_t)l, 0, 0);
#else
  unsigned lo = (unsigned)(uintptr_t)l;
  asm volatile("global_load_async_to_lds_b128 %0, %1, off"
               :: "v"(lo), "v"(g) : "memory");
#endif
}

template <int N>
static __device__ __forceinline__ void wait_async() {
#if defined(ASYNC_WAIT_BUILTIN)
  __builtin_amdgcn_s_wait_asynccnt(N);
#else
  if (N == 0) asm volatile("s_wait_asynccnt 0x0" ::: "memory");
  else        asm volatile("s_wait_asynccnt 0x3" ::: "memory");
#endif
}

// ---------------- Workspace layout (bytes) ----------------
#define WB_INPROJ 0u
#define WB_WSUM   65536u
#define WB_TITOUT 589824u
#define WB_QKV    1114112u
#define WB_AO     2686976u
#define WB_F1     3211264u
#define WB_F2     5308416u
#define XN_OFF    7405568u
#define HF32_OFF  8454144u          // also reused as hfin at the end
#define HB16_OFF  25231360u
#define T1B16_OFF 33619968u         // reused as attention-output bf16
#define H2F32_OFF 42008576u
#define HNB16_OFF 58785792u
#define QKVB_OFF  67174400u
#define GB16_OFF  92340224u
#define MEANS_OFF 125894656u
#define CMSO_OFF  127991808u
#define TARG_OFF  127993856u
#define ERR_OFF   128010240u
#define QM_OFF    128012288u
#define GSC_OFF   128014336u

// ---------------- Output layout (elements) ----------------
#define OUT_PRED 0
#define OUT_W    8192
#define OUT_NS   270336
#define OUT_NB   271872
#define OUT_NC   273408
#define OUT_NT   273411

// =======================================================================
// Small conversion kernels
// =======================================================================
__global__ void k_cvt_bf16(const float* __restrict__ in, __bf16* __restrict__ out) {
  int i = blockIdx.x * blockDim.x + threadIdx.x;
  out[i] = (__bf16)in[i];
}

__global__ void k_wsum_bf16(const float* __restrict__ a, const float* __restrict__ b,
                            __bf16* __restrict__ out) {
  int i = blockIdx.x * blockDim.x + threadIdx.x;
  out[i] = (__bf16)(a[i] + b[i]);
}

// =======================================================================
// LayerNorm over F=64 (input) -> bf16
// =======================================================================
__global__ void k_ln64(const float* __restrict__ x, const float* __restrict__ w,
                       const float* __restrict__ b, __bf16* __restrict__ out) {
  __shared__ float sh[64];
  int row = blockIdx.x, t = threadIdx.x;
  float v = x[(size_t)row * Fz + t];
  sh[t] = v; __syncthreads();
  for (int s = 32; s > 0; s >>= 1) { if (t < s) sh[t] += sh[t + s]; __syncthreads(); }
  float mean = sh[0] * (1.0f / Fz);
  __syncthreads();
  float dv = v - mean;
  sh[t] = dv * dv; __syncthreads();
  for (int s = 32; s > 0; s >>= 1) { if (t < s) sh[t] += sh[t + s]; __syncthreads(); }
  float var = sh[0] * (1.0f / Fz);
  float r = rsqrtf(var + 1e-5f);
  out[(size_t)row * Fz + t] = (__bf16)(dv * r * w[t] + b[t]);
}

// =======================================================================
// LayerNorm over D=512 -> optional bf16 / f32 outputs
// =======================================================================
__global__ void k_ln512(const float* __restrict__ x, const float* __restrict__ w,
                        const float* __restrict__ b, __bf16* __restrict__ outB,
                        float* __restrict__ outF) {
  __shared__ float sh[256];
  int row = blockIdx.x, t = threadIdx.x;
  const float* xr = x + (size_t)row * Dz;
  float v0 = xr[t], v1 = xr[t + 256];
  sh[t] = v0 + v1; __syncthreads();
  for (int s = 128; s > 0; s >>= 1) { if (t < s) sh[t] += sh[t + s]; __syncthreads(); }
  float mean = sh[0] * (1.0f / Dz);
  __syncthreads();
  float d0 = v0 - mean, d1 = v1 - mean;
  sh[t] = d0 * d0 + d1 * d1; __syncthreads();
  for (int s = 128; s > 0; s >>= 1) { if (t < s) sh[t] += sh[t + s]; __syncthreads(); }
  float var = sh[0] * (1.0f / Dz);
  float r = rsqrtf(var + 1e-5f);
  float y0 = d0 * r * w[t] + b[t];
  float y1 = d1 * r * w[t + 256] + b[t + 256];
  size_t o = (size_t)row * Dz + t;
  if (outB) { outB[o] = (__bf16)y0; outB[o + 256] = (__bf16)y1; }
  if (outF) { outF[o] = y0;         outF[o + 256] = y1; }
}

// =======================================================================
// LDS-staged, double-buffered, async-copy bf16 WMMA GEMM
//   C[M,N] = A[M,K] @ W[N,K]^T (+bias +bias2 +res), ACT: 0 none, 1 GELU
// Block: 256 threads = 8 waves; tile 128(M) x 64(N); k-step 32.
// LDS pitches padded to 40 elements (80B = 20-bank stride) to destagger.
// =======================================================================
#define APITCH 40
#define BPITCH 40

template <int ACT>
__global__ __launch_bounds__(256) void k_gemm_bf16(
    const __bf16* __restrict__ A, const __bf16* __restrict__ W,
    const float* __restrict__ bias, const float* __restrict__ bias2,
    const float* __restrict__ res, float* __restrict__ outF,
    __bf16* __restrict__ outB, int M, int N, int K) {
  __shared__ __bf16 smA[2][128 * APITCH];
  __shared__ __bf16 smB[2][64 * BPITCH];
  const int tid = threadIdx.x;
  const int wv = tid >> 5;
  const int lane = tid & 31;
  const int half = lane >> 4;
  const int ln = lane & 15;
  const int mblk = blockIdx.y * 128;
  const int c0 = blockIdx.x * 64;

  // staging coordinates: 16B chunks; A: 128 rows x 4 chunks (2/thread),
  // B: 64 rows x 4 chunks (1/thread)
  const int srow = tid >> 2;        // 0..63
  const int sch = (tid & 3) * 8;    // element offset within 32-k slab

  v8f acc[4];
#pragma unroll
  for (int nt = 0; nt < 4; ++nt) acc[nt] = (v8f)(0.0f);

  const int NS = K >> 5;

  // prologue: stage k-step 0 into buffer 0
  {
    async_ld_b128(A + (size_t)(mblk + srow) * K + sch, &smA[0][srow * APITCH + sch]);
    async_ld_b128(A + (size_t)(mblk + 64 + srow) * K + sch,
                  &smA[0][(64 + srow) * APITCH + sch]);
    async_ld_b128(W + (size_t)(c0 + srow) * K + sch, &smB[0][srow * BPITCH + sch]);
  }

  for (int s = 0; s < NS; ++s) {
    const int buf = s & 1;
    if (s + 1 < NS) {
      const int k0 = (s + 1) << 5;
      async_ld_b128(A + (size_t)(mblk + srow) * K + k0 + sch,
                    &smA[buf ^ 1][srow * APITCH + sch]);
      async_ld_b128(A + (size_t)(mblk + 64 + srow) * K + k0 + sch,
                    &smA[buf ^ 1][(64 + srow) * APITCH + sch]);
      async_ld_b128(W + (size_t)(c0 + srow) * K + k0 + sch,
                    &smB[buf ^ 1][srow * BPITCH + sch]);
      wait_async<3>();   // async loads complete in order: current buffer done
    } else {
      wait_async<0>();
    }
    __syncthreads();

    const __bf16* arow = &smA[buf][(wv * 16 + ln) * APITCH];
    v16bf af = mk16(*(const v8bf*)(arow + half * 8),
                    *(const v8bf*)(arow + half * 8 + 16));
#pragma unroll
    for (int nt = 0; nt < 4; ++nt) {
      const __bf16* br = &smB[buf][(nt * 16 + ln) * BPITCH + half * 8];
      v16bf bf = mk16(*(const v8bf*)br, *(const v8bf*)(br + 16));
      acc[nt] = __builtin_amdgcn_wmma_f32_16x16x32_bf16(
          false, af, false, bf, (short)0, acc[nt], false, false);
    }
    __syncthreads();   // protect buf from being re-staged while still read
  }

  const int m0 = mblk + wv * 16;
#pragma unroll
  for (int nt = 0; nt < 4; ++nt) {
    int col = c0 + nt * 16 + ln;
    float bv = (bias ? bias[col] : 0.0f) + (bias2 ? bias2[col] : 0.0f);
#pragma unroll
    for (int r = 0; r < 8; ++r) {
      int row = m0 + r + half * 8;
      size_t idx = (size_t)row * N + col;
      float v = acc[nt][r] + bv;
      if (res) v += res[idx];
      if (ACT == 1) v = 0.5f * v * (1.0f + erff(v * 0.70710678118654752f));
      if (outF) outF[idx] = v;
      if (outB) outB[idx] = (__bf16)v;
    }
  }
}

// =======================================================================
// Flash attention, causal, H=8, DH=64, bf16 inputs (packed qkv rows of 1536)
// grid = B*H*(T/128), block = 256 (8 waves, 16 q-rows each)
// K tile staged with async global->LDS; padded LDS pitches.
// =======================================================================
#define KPITCH 72   // 144B rows, 36-bank stride
#define VPITCH 40   // 80B rows
#define PPITCH 40   // 80B rows

__global__ __launch_bounds__(256) void k_attn(const __bf16* __restrict__ qkv,
                                              __bf16* __restrict__ ao) {
  const int bidx = blockIdx.x;
  const int qblk = bidx & 7;
  const int hh = (bidx >> 3) & 7;
  const int b = bidx >> 6;
  const int tid = threadIdx.x;
  const int wv = tid >> 5;
  const int lane = tid & 31;
  const int half = lane >> 4;
  const int ln = lane & 15;

  __shared__ __bf16 kt[32 * KPITCH];       // K tile, row-major [key][dh]
  __shared__ __bf16 vt[64 * VPITCH];       // V tile, transposed [dh][key]
  __shared__ __bf16 pl[8 * 16 * PPITCH];   // per-wave P tiles [m][key]

  const int qbase = qblk * 128 + wv * 16;
  const __bf16* qrow = qkv + ((size_t)(b * Tz + qbase + ln)) * (3 * Dz) + hh * DHz;
  v16bf qf0 = mk16(*(const v8bf*)(qrow + half * 8),
                   *(const v8bf*)(qrow + half * 8 + 16));
  v16bf qf1 = mk16(*(const v8bf*)(qrow + 32 + half * 8),
                   *(const v8bf*)(qrow + 32 + half * 8 + 16));

  v8f o[4];
#pragma unroll
  for (int nt = 0; nt < 4; ++nt) o[nt] = (v8f)(0.0f);
  float mrow[8], lrow[8];
#pragma unroll
  for (int r = 0; r < 8; ++r) { mrow[r] = -1e30f; lrow[r] = 0.0f; }

  const int jmax = qblk * 128 + 128;
  for (int j0 = 0; j0 < jmax; j0 += 32) {
    // cooperative K/V tile staging: K async, V transposed via scalar stores
    {
      int key = tid >> 3;
      int dh0 = (tid & 7) * 8;
      const __bf16* kr = qkv + ((size_t)(b * Tz + j0 + key)) * (3 * Dz) + Dz + hh * DHz + dh0;
      async_ld_b128(kr, &kt[key * KPITCH + dh0]);
      const __bf16* vr = qkv + ((size_t)(b * Tz + j0 + key)) * (3 * Dz) + 2 * Dz + hh * DHz + dh0;
      v8bf vv = *(const v8bf*)vr;
#pragma unroll
      for (int i = 0; i < 8; ++i) vt[(dh0 + i) * VPITCH + key] = vv[i];
      wait_async<0>();
    }
    __syncthreads();

    // S = Q K^T for two 16-key groups
    v8f s0 = (v8f)(0.0f), s1 = (v8f)(0.0f);
    {
      const __bf16* kb = &kt[(0 * 16 + ln) * KPITCH + half * 8];
      v16bf b0 = mk16(*(const v8bf*)(kb), *(const v8bf*)(kb + 16));
      v16bf b1 = mk16(*(const v8bf*)(kb + 32), *(const v8bf*)(kb + 48));
      s0 = __builtin_amdgcn_wmma_f32_16x16x32_bf16(false, qf0, false, b0, (short)0, s0, false, false);
      s0 = __builtin_amdgcn_wmma_f32_16x16x32_bf16(false, qf1, false, b1, (short)0, s0, false, false);
    }
    {
      const __bf16* kb = &kt[(1 * 16 + ln) * KPITCH + half * 8];
      v16bf b0 = mk16(*(const v8bf*)(kb), *(const v8bf*)(kb + 16));
      v16bf b1 = mk16(*(const v8bf*)(kb + 32), *(const v8bf*)(kb + 48));
      s1 = __builtin_amdgcn_wmma_f32_16x16x32_bf16(false, qf0, false, b0, (short)0, s1, false, false);
      s1 = __builtin_amdgcn_wmma_f32_16x16x32_bf16(false, qf1, false, b1, (short)0, s1, false, false);
    }

    // online softmax update per row
#pragma unroll
    for (int r = 0; r < 8; ++r) {
      int row = qbase + r + half * 8;
      float v0 = s0[r] * 0.125f;
      float v1 = s1[r] * 0.125f;
      if (j0 + ln > row) v0 = -1e30f;
      if (j0 + 16 + ln > row) v1 = -1e30f;
      float mx = fmaxf(v0, v1);
      mx = fmaxf(mx, __shfl_xor(mx, 1, 32));
      mx = fmaxf(mx, __shfl_xor(mx, 2, 32));
      mx = fmaxf(mx, __shfl_xor(mx, 4, 32));
      mx = fmaxf(mx, __shfl_xor(mx, 8, 32));
      float mnew = fmaxf(mrow[r], mx);
      float alpha = __expf(mrow[r] - mnew);
      float p0 = __expf(v0 - mnew);
      float p1 = __expf(v1 - mnew);
      float rs = p0 + p1;
      rs += __shfl_xor(rs, 1, 32);
      rs += __shfl_xor(rs, 2, 32);
      rs += __shfl_xor(rs, 4, 32);
      rs += __shfl_xor(rs, 8, 32);
      lrow[r] = lrow[r] * alpha + rs;
      mrow[r] = mnew;
#pragma unroll
      for (int nt = 0; nt < 4; ++nt) o[nt][r] *= alpha;
      pl[wv * (16 * PPITCH) + (r + half * 8) * PPITCH + ln] = (__bf16)p0;
      pl[wv * (16 * PPITCH) + (r + half * 8) * PPITCH + 16 + ln] = (__bf16)p1;
    }
    __syncthreads();

    // O += P @ V
    {
      const __bf16* pp = &pl[wv * (16 * PPITCH) + ln * PPITCH + half * 8];
      v16bf pf = mk16(*(const v8bf*)(pp), *(const v8bf*)(pp + 16));
#pragma unroll
      for (int nt = 0; nt < 4; ++nt) {
        const __bf16* vb = &vt[(nt * 16 + ln) * VPITCH + half * 8];
        v16bf bf = mk16(*(const v8bf*)(vb), *(const v8bf*)(vb + 16));
        o[nt] = __builtin_amdgcn_wmma_f32_16x16x32_bf16(
            false, pf, false, bf, (short)0, o[nt], false, false);
      }
    }
    __syncthreads();
  }

  // epilogue: normalize + write bf16 (already merged-head layout [b,t,h*64+dh])
#pragma unroll
  for (int r = 0; r < 8; ++r) {
    int row = qbase + r + half * 8;
    float inv = 1.0f / lrow[r];
#pragma unroll
    for (int nt = 0; nt < 4; ++nt) {
      ao[((size_t)(b * Tz + row)) * Dz + hh * DHz + nt * 16 + ln] = (__bf16)(o[nt][r] * inv);
    }
  }
}

// =======================================================================
// Head + means
// =======================================================================
__global__ void k_head(const float* __restrict__ h, const float* __restrict__ hw,
                       const float* __restrict__ hb, float* __restrict__ out) {
  int i = blockIdx.x * blockDim.x + threadIdx.x;  // 0..8191
  const float* r = h + (size_t)i * Dz;
  float s = hb[0];
  for (int j = 0; j < Dz; ++j) s += r[j] * hw[j];
  out[i] = s;
}

__global__ void k_means(const float* __restrict__ h, float* __restrict__ means) {
  int idx = blockIdx.x * blockDim.x + threadIdx.x;  // t*512+d
  float s = 0.0f;
  for (int b = 0; b < Bz; ++b) s += h[(size_t)b * Tz * Dz + idx];
  means[idx] = s * (1.0f / Bz);
}

// =======================================================================
// Titans memory update
// =======================================================================
__global__ void k_titans_target(const float* __restrict__ h, float* __restrict__ target) {
  int idx = blockIdx.x * blockDim.x + threadIdx.x;  // b*512+d
  int b = idx >> 9, d = idx & 511;
  const float* base = h + (size_t)b * Tz * Dz + d;
  float s = 0.0f;
  for (int t = 0; t < Tz; ++t) s += base[(size_t)t * Dz];
  target[idx] = s * (1.0f / Tz);
}

__global__ void k_titans_errq(const float* __restrict__ h, const float* __restrict__ tW,
                              const float* __restrict__ target, float* __restrict__ err,
                              float* __restrict__ qmean) {
  int d = blockIdx.x * blockDim.x + threadIdx.x;  // 0..511
  float qm = 0.0f, e = 0.0f;
  const float* wrow = tW + (size_t)d * Dz;
  for (int b = 0; b < Bz; ++b) {
    const float* q = h + ((size_t)b * Tz + (Tz - 1)) * Dz;
    qm += q[d];
    float dot = 0.0f;
    for (int j = 0; j < Dz; ++j) dot += q[j] * wrow[j];
    e += target[b * Dz + d] - dot;
  }
  err[d] = e * (1.0f / Bz);
  qmean[d] = qm * (1.0f / Bz);
}

__global__ void k_titans_norm(const float* __restrict__ err, const float* __restrict__ qm,
                              float* __restrict__ gscale) {
  __shared__ float sh[256];
  int t = threadIdx.x;
  float a = err[t] * err[t] + err[t + 256] * err[t + 256];
  sh[t] = a; __syncthreads();
  for (int s = 128; s > 0; s >>= 1) { if (t < s) sh[t] += sh[t + s]; __syncthreads(); }
  float se = sh[0]; __syncthreads();
  float b = qm[t] * qm[t] + qm[t + 256] * qm[t + 256];
  sh[t] = b; __syncthreads();
  for (int s = 128; s > 0; s >>= 1) { if (t < s) sh[t] += sh[t + s]; __syncthreads(); }
  if (t == 0) {
    float gn = sqrtf(se * sh[0]);
    gscale[0] = 0.01f * ((gn > 0.1f) ? (0.1f / gn) : 1.0f);
  }
}

__global__ void k_titans_neww(const float* __restrict__ tW, const float* __restrict__ err,
                              const float* __restrict__ qm, const float* __restrict__ gscale,
                              float* __restrict__ out) {
  int idx = blockIdx.x * blockDim.x + threadIdx.x;  // 0..262143
  int i = idx >> 9, j = idx & 511;
  out[idx] = tW[idx] + gscale[0] * err[i] * qm[j];
}

// =======================================================================
// CMS: summary projection (cms_out) + sequential scan over T
// =======================================================================
__global__ void k_cms_gemv(const float* __restrict__ summ, const float* __restrict__ combW,
                           const float* __restrict__ combB, float* __restrict__ out) {
  int d = blockIdx.x * blockDim.x + threadIdx.x;  // 0..511
  float v = combB[d];
  const float* wrow = combW + (size_t)d * (Lz * Dz);
  for (int j = 0; j < Lz * Dz; ++j) v += summ[j] * wrow[j];
  out[d] = v;
}

__global__ void k_cms_scan(const float* __restrict__ means, const float* __restrict__ summ_in,
                           const float* __restrict__ buf_in, const int* __restrict__ cnt_in,
                           const int* __restrict__ step_in, const float* __restrict__ gateW,
                           const float* __restrict__ gateB, float* __restrict__ out) {
  __shared__ float accs[Dz];
  const int d = threadIdx.x;  // 0..511
  const int per[3] = {16, 256, 4096};
  const float lr[3] = {0.01f, 0.001f, 0.0001f};
  float summ[3], bs[3];
  int cnt[3];
#pragma unroll
  for (int l = 0; l < Lz; ++l) {
    summ[l] = summ_in[l * Dz + d];
    bs[l] = buf_in[l * Dz + d];
    cnt[l] = cnt_in[l];
  }
  for (int t = 0; t < Tz; ++t) {
    float m = means[(size_t)t * Dz + d];
#pragma unroll
    for (int l = 0; l < Lz; ++l) {
      bs[l] += m;
      cnt[l] += 1;
      if (cnt[l] >= per[l]) {  // block-uniform condition
        float acc = bs[l] / (float)cnt[l];
        __syncthreads();
        accs[d] = acc;
        __syncthreads();
        float gsum = gateB[l * Dz + d];
        const float* grow = gateW + ((size_t)l * Dz + d) * Dz;
        for (int j = 0; j < 512; ++j) gsum += grow[j] * accs[j];
        float g = 1.0f / (1.0f + __expf(-gsum));
        summ[l] = (1.0f - lr[l]) * summ[l] + lr[l] * g * acc;
        bs[l] = 0.0f;
        cnt[l] = 0;
      }
    }
  }
#pragma unroll
  for (int l = 0; l < Lz; ++l) {
    out[OUT_NS + l * Dz + d] = summ[l];
    out[OUT_NB + l * Dz + d] = bs[l];
  }
  if (d < Lz) {
    out[OUT_NC + d] = (float)cnt[d];
    out[OUT_NT + d] = (float)(step_in[d] + Tz);
  }
}

// =======================================================================
// Host launcher
// =======================================================================
extern "C" void kernel_launch(void* const* d_in, const int* in_sizes, int n_in,
                              void* d_out, int out_size, void* d_ws, size_t ws_size,
                              hipStream_t stream) {
  (void)in_sizes; (void)n_in; (void)out_size; (void)ws_size;
  const float* x         = (const float*)d_in[0];
  const float* titans_W  = (const float*)d_in[1];
  const float* cms_summ  = (const float*)d_in[2];
  const float* cms_buf   = (const float*)d_in[3];
  const int*   cms_cnt   = (const int*)d_in[4];
  const int*   cms_step  = (const int*)d_in[5];
  const float* in_nw     = (const float*)d_in[6];
  const float* in_nb     = (const float*)d_in[7];
  const float* in_projW  = (const float*)d_in[8];
  const float* in_projb  = (const float*)d_in[9];
  const float* W_base    = (const float*)d_in[10];
  const float* tit_outW  = (const float*)d_in[11];
  const float* tit_outb  = (const float*)d_in[12];
  const float* gateW     = (const float*)d_in[13];
  const float* gateB     = (const float*)d_in[14];
  const float* combW     = (const float*)d_in[15];
  const float* combB     = (const float*)d_in[16];
  const float* n1w       = (const float*)d_in[17];
  const float* n1b       = (const float*)d_in[18];
  const float* qkvW      = (const float*)d_in[19];
  const float* qkvb      = (const float*)d_in[20];
  const float* aoW       = (const float*)d_in[21];
  const float* aob       = (const float*)d_in[22];
  const float* n2w       = (const float*)d_in[23];
  const float* n2b       = (const float*)d_in[24];
  const float* f1W       = (const float*)d_in[25];
  const float* f1b       = (const float*)d_in[26];
  const float* f2W       = (const float*)d_in[27];
  const float* f2b       = (const float*)d_in[28];
  const float* fnw       = (const float*)d_in[29];
  const float* fnb       = (const float*)d_in[30];
  const float* headW     = (const float*)d_in[31];
  const float* headb     = (const float*)d_in[32];
  float* out = (float*)d_out;
  char* ws = (char*)d_ws;

  __bf16* wb_inproj = (__bf16*)(ws + WB_INPROJ);
  __bf16* wb_wsum   = (__bf16*)(ws + WB_WSUM);
  __bf16* wb_titout = (__bf16*)(ws + WB_TITOUT);
  __bf16* wb_qkv    = (__bf16*)(ws + WB_QKV);
  __bf16* wb_ao     = (__bf16*)(ws + WB_AO);
  __bf16* wb_f1     = (__bf16*)(ws + WB_F1);
  __bf16* wb_f2     = (__bf16*)(ws + WB_F2);
  __bf16* xn        = (__bf16*)(ws + XN_OFF);
  float*  hF32      = (float*)(ws + HF32_OFF);
  __bf16* hB16      = (__bf16*)(ws + HB16_OFF);
  __bf16* t1B16     = (__bf16*)(ws + T1B16_OFF);
  float*  h2F32     = (float*)(ws + H2F32_OFF);
  __bf16* hnB16     = (__bf16*)(ws + HNB16_OFF);
  __bf16* qkvB16    = (__bf16*)(ws + QKVB_OFF);
  __bf16* gB16      = (__bf16*)(ws + GB16_OFF);
  float*  means     = (float*)(ws + MEANS_OFF);
  float*  cmsout    = (float*)(ws + CMSO_OFF);
  float*  target    = (float*)(ws + TARG_OFF);
  float*  errv      = (float*)(ws + ERR_OFF);
  float*  qmv       = (float*)(ws + QM_OFF);
  float*  gsc       = (float*)(ws + GSC_OFF);

  // weight conversion
  hipLaunchKernelGGL(k_cvt_bf16, dim3(128), dim3(256), 0, stream, in_projW, wb_inproj);
  hipLaunchKernelGGL(k_wsum_bf16, dim3(1024), dim3(256), 0, stream, W_base, titans_W, wb_wsum);
  hipLaunchKernelGGL(k_cvt_bf16, dim3(1024), dim3(256), 0, stream, tit_outW, wb_titout);
  hipLaunchKernelGGL(k_cvt_bf16, dim3(3072), dim3(256), 0, stream, qkvW, wb_qkv);
  hipLaunchKernelGGL(k_cvt_bf16, dim3(1024), dim3(256), 0, stream, aoW, wb_ao);
  hipLaunchKernelGGL(k_cvt_bf16, dim3(4096), dim3(256), 0, stream, f1W, wb_f1);
  hipLaunchKernelGGL(k_cvt_bf16, dim3(4096), dim3(256), 0, stream, f2W, wb_f2);

  // input LN + in_proj
  hipLaunchKernelGGL(k_ln64, dim3(BT), dim3(64), 0, stream, x, in_nw, in_nb, xn);
  hipLaunchKernelGGL(k_gemm_bf16<0>, dim3(Dz / 64, BT / 128), dim3(256), 0, stream,
                     xn, wb_inproj, in_projb, (const float*)nullptr, (const float*)nullptr,
                     hF32, hB16, BT, Dz, Fz);

  // titans weight update (uses pre-residual h)
  hipLaunchKernelGGL(k_titans_target, dim3(16), dim3(256), 0, stream, hF32, target);
  hipLaunchKernelGGL(k_titans_errq, dim3(2), dim3(256), 0, stream, hF32, titans_W, target, errv, qmv);
  hipLaunchKernelGGL(k_titans_norm, dim3(1), dim3(256), 0, stream, errv, qmv, gsc);
  hipLaunchKernelGGL(k_titans_neww, dim3(1024), dim3(256), 0, stream, titans_W, errv, qmv, gsc,
                     out + OUT_W);

  // cms summary projection (bias2 for tit GEMM epilogue)
  hipLaunchKernelGGL(k_cms_gemv, dim3(2), dim3(256), 0, stream, cms_summ, combW, combB, cmsout);

  // tit = h @ Wsum^T @ tit_out^T ; h2 = h + tit + cms_out
  hipLaunchKernelGGL(k_gemm_bf16<0>, dim3(Dz / 64, BT / 128), dim3(256), 0, stream,
                     hB16, wb_wsum, (const float*)nullptr, (const float*)nullptr,
                     (const float*)nullptr, (float*)nullptr, t1B16, BT, Dz, Dz);
  hipLaunchKernelGGL(k_gemm_bf16<0>, dim3(Dz / 64, BT / 128), dim3(256), 0, stream,
                     t1B16, wb_titout, tit_outb, cmsout, hF32, h2F32, (__bf16*)nullptr,
                     BT, Dz, Dz);

  // attention
  hipLaunchKernelGGL(k_ln512, dim3(BT), dim3(256), 0, stream, h2F32, n1w, n1b, hnB16,
                     (float*)nullptr);
  hipLaunchKernelGGL(k_gemm_bf16<0>, dim3((3 * Dz) / 64, BT / 128), dim3(256), 0, stream,
                     hnB16, wb_qkv, qkvb, (const float*)nullptr, (const float*)nullptr,
                     (float*)nullptr, qkvB16, BT, 3 * Dz, Dz);
  hipLaunchKernelGGL(k_attn, dim3(Bz * Hz * (Tz / 128)), dim3(256), 0, stream, qkvB16, t1B16);
  hipLaunchKernelGGL(k_gemm_bf16<0>, dim3(Dz / 64, BT / 128), dim3(256), 0, stream,
                     t1B16, wb_ao, aob, (const float*)nullptr, h2F32, h2F32, (__bf16*)nullptr,
                     BT, Dz, Dz);

  // FFN
  hipLaunchKernelGGL(k_ln512, dim3(BT), dim3(256), 0, stream, h2F32, n2w, n2b, hnB16,
                     (float*)nullptr);
  hipLaunchKernelGGL(k_gemm_bf16<1>, dim3(DFFz / 64, BT / 128), dim3(256), 0, stream,
                     hnB16, wb_f1, f1b, (const float*)nullptr, (const float*)nullptr,
                     (float*)nullptr, gB16, BT, DFFz, Dz);
  hipLaunchKernelGGL(k_gemm_bf16<0>, dim3(Dz / 64, BT / 128), dim3(256), 0, stream,
                     gB16, wb_f2, f2b, (const float*)nullptr, h2F32, h2F32, (__bf16*)nullptr,
                     BT, Dz, DFFz);

  // final LN (into hF32, which is free now), head, means, CMS scan
  hipLaunchKernelGGL(k_ln512, dim3(BT), dim3(256), 0, stream, h2F32, fnw, fnb,
                     (__bf16*)nullptr, hF32);
  hipLaunchKernelGGL(k_head, dim3(BT / 256), dim3(256), 0, stream, hF32, headW, headb,
                     out + OUT_PRED);
  hipLaunchKernelGGL(k_means, dim3((Tz * Dz) / 256), dim3(256), 0, stream, hF32, means);
  hipLaunchKernelGGL(k_cms_scan, dim3(1), dim3(Dz), 0, stream, means, cms_summ, cms_buf,
                     cms_cnt, cms_step, gateW, gateB, out);
}
